// HeteroGAE_AA_Decoder_33818572489426
// MI455X (gfx1250) — matrix operand order repaired and checked by reading, not compiled, not run on hardware
//
#include <hip/hip_runtime.h>

// ---------------------------------------------------------------------------
// Types for CDNA5 WMMA (wave32, 16x16x32 bf16 -> f32)
// ---------------------------------------------------------------------------
typedef __bf16 bf16_t;
typedef __attribute__((ext_vector_type(16))) __bf16 v16bf;
typedef __attribute__((ext_vector_type(8)))  float  v8f;

__device__ __forceinline__ float apply_act(float x, int act) {
    if (act == 1) return 0.5f * x * (1.0f + erff(x * 0.70710678118654752440f)); // exact GELU
    if (act == 2) return tanhf(x);
    return x;
}

__device__ __forceinline__ v16bf load_afrag(const bf16_t* __restrict__ Arow) {
    // ISA 16-bit A 16x32 layout: element e=(vg,h): K = (vg<4 ? 2vg : 16+2(vg-4)) + h
    v16bf a;
#pragma unroll
    for (int vg = 0; vg < 8; ++vg) {
        const int kk = (vg < 4) ? (2 * vg) : (16 + 2 * (vg - 4));
        a[2 * vg]     = Arow[kk];
        a[2 * vg + 1] = Arow[kk + 1];
    }
    return a;
}

// ---------------------------------------------------------------------------
// Generic tall-skinny GEMM:  C[N,F] = act( A[N,K]@W[K,F] (+ A2@W2) + bias )
// A/A2/W/W2 bf16, accumulate f32 via v_wmma_f32_16x16x32_bf16.
// K multiple of 32, F multiple of 64 (buffers zero-padded).
// Block = 256 threads = 8 waves stacked in M; each wave computes a 16x64
// output tile (4 WMMA accumulators sharing one A fragment). All fragment
// loads for a K-step are issued before the WMMA chain so the matrix pipe
// overlaps with outstanding loads (partial s_wait_loadcnt). DUAL is a
// compile-time flag (no in-loop branch -> no accumulator copies).
// Grid: x = ceil(N/128), y = F/64.
// ---------------------------------------------------------------------------
template <bool DUAL>
__global__ __launch_bounds__(256)
void wmma_gemm_tpl(const bf16_t* __restrict__ A,  const bf16_t* __restrict__ A2,
                   const bf16_t* __restrict__ W,  const bf16_t* __restrict__ W2,
                   const float*  __restrict__ bias, int biasLen,
                   float* __restrict__ C, bf16_t* __restrict__ Cbf,
                   int N, int K, int F, int act)
{
    const int lane = threadIdx.x & 31;
    const int wave = threadIdx.x >> 5;
    const long mTile = (long)blockIdx.x * 128 + wave * 16;
    const int  nTile = blockIdx.y * 64;

    // A-fragment row for this lane (lanes 0-15 & 16-31 both carry rows
    // M=0..15; lane-half selects the K sub-block).
    long rA = mTile + (lane & 15);
    if (rA >= N) rA = N - 1;               // clamp; stores are masked below
    const int kb = (lane >> 4) * 8;        // K base offset for this lane half

    v8f acc[4] = {v8f{}, v8f{}, v8f{}, v8f{}};
    for (int k0 = 0; k0 < K; k0 += 32) {
        // ---- issue ALL loads for this K-step first ----
        const v16bf a = load_afrag(A + (size_t)rA * K + k0 + kb);
        const bf16_t* Wrow = W + (size_t)(k0 + kb) * F + nTile + (lane & 15);
        v16bf b[4];
#pragma unroll
        for (int t = 0; t < 4; ++t)
#pragma unroll
            for (int s = 0; s < 16; ++s) b[t][s] = Wrow[(size_t)(s < 8 ? s : s + 8) * F + 16 * t];

        if constexpr (DUAL) {
            const v16bf a2 = load_afrag(A2 + (size_t)rA * K + k0 + kb);
            const bf16_t* W2row = W2 + (size_t)(k0 + kb) * F + nTile + (lane & 15);
            v16bf b2[4];
#pragma unroll
            for (int t = 0; t < 4; ++t)
#pragma unroll
                for (int s = 0; s < 16; ++s) b2[t][s] = W2row[(size_t)(s < 8 ? s : s + 8) * F + 16 * t];

#pragma unroll
            for (int t = 0; t < 4; ++t)
                acc[t] = __builtin_amdgcn_wmma_f32_16x16x32_bf16(false, a, false, b[t],
                                                                 (short)0, acc[t], false, false);
#pragma unroll
            for (int t = 0; t < 4; ++t)
                acc[t] = __builtin_amdgcn_wmma_f32_16x16x32_bf16(false, a2, false, b2[t],
                                                                 (short)0, acc[t], false, false);
        } else {
#pragma unroll
            for (int t = 0; t < 4; ++t)
                acc[t] = __builtin_amdgcn_wmma_f32_16x16x32_bf16(false, a, false, b[t],
                                                                 (short)0, acc[t], false, false);
        }
    }

    // ---- epilogue: D layout lane -> N=lane&15, VGPR vg -> M = vg + 8*(lane>>4)
    const int n   = lane & 15;
    const int mhi = (lane >> 4) * 8;
#pragma unroll
    for (int t = 0; t < 4; ++t) {
        const int fcol = nTile + 16 * t + n;
        const float bv = (bias && fcol < biasLen) ? bias[fcol] : 0.0f;
#pragma unroll
        for (int vg = 0; vg < 8; ++vg) {
            const long r = mTile + mhi + vg;
            if (r < N) {
                float v = apply_act(acc[t][vg] + bv, act);
                if (C)   C  [(size_t)r * F + fcol] = v;
                if (Cbf) Cbf[(size_t)r * F + fcol] = (bf16_t)v;
            }
        }
    }
}

// ---------------------------------------------------------------------------
// Elementwise / graph kernels
// ---------------------------------------------------------------------------
__global__ void zero_f32_kernel(float* __restrict__ p, long n) {
    long i = (long)blockIdx.x * blockDim.x + threadIdx.x;
    if (i < n) p[i] = 0.0f;
}

__global__ void concat_kernel(const float* __restrict__ xres, const float* __restrict__ pos,
                              float* __restrict__ inzf, bf16_t* __restrict__ hbf, int N) {
    long idx = (long)blockIdx.x * blockDim.x + threadIdx.x;
    long total = (long)N * 288;
    if (idx >= total) return;
    int n = (int)(idx / 288), f = (int)(idx % 288);
    float v = 0.0f;
    if (f < 10)       v = xres[(size_t)n * 10 + f];
    else if (f < 266) v = pos [(size_t)n * 256 + (f - 10)];
    inzf[idx] = v;
    hbf [idx] = (bf16_t)v;
}

__global__ void degree_kernel(const int* __restrict__ dst, float* __restrict__ deg, int E) {
    long i = (long)blockIdx.x * blockDim.x + threadIdx.x;
    if (i < E) atomicAdd(&deg[dst[i]], 1.0f);
}

__global__ void scatter_add_kernel(const float* __restrict__ h, const int* __restrict__ src,
                                   const int* __restrict__ dst, float* __restrict__ aggr,
                                   long total, int Kin, int strideIn, int strideOut) {
    long idx = (long)blockIdx.x * blockDim.x + threadIdx.x;
    if (idx >= total) return;
    int e = (int)(idx / Kin);
    int f = (int)(idx % Kin);
    atomicAdd(&aggr[(size_t)dst[e] * strideOut + f], h[(size_t)src[e] * strideIn + f]);
}

__global__ void aggr_finalize_kernel(const float* __restrict__ aggrf, const float* __restrict__ deg,
                                     bf16_t* __restrict__ aggrbf, int N, int Kin, int Kpad) {
    long idx = (long)blockIdx.x * blockDim.x + threadIdx.x;
    long total = (long)N * Kpad;
    if (idx >= total) return;
    int n = (int)(idx / Kpad), f = (int)(idx % Kpad);
    float v = 0.0f;
    if (f < Kin) v = aggrf[idx] / fmaxf(deg[n], 1.0f);
    aggrbf[idx] = (bf16_t)v;
}

// ---- GraphNorm: per-feature column reductions (one block per feature) ------
__global__ __launch_bounds__(256)
void colsum_kernel(const float* __restrict__ x, float* __restrict__ mean, int N, int F) {
    __shared__ float sdata[256];
    const int f = blockIdx.x;
    float s = 0.0f;
    for (int r = threadIdx.x; r < N; r += 256) s += x[(size_t)r * F + f];
    sdata[threadIdx.x] = s;
    __syncthreads();
    for (int off = 128; off > 0; off >>= 1) {
        if (threadIdx.x < off) sdata[threadIdx.x] += sdata[threadIdx.x + off];
        __syncthreads();
    }
    if (threadIdx.x == 0) mean[f] = sdata[0] / (float)N;
}

__global__ __launch_bounds__(256)
void colvar_kernel(const float* __restrict__ x, const float* __restrict__ mean,
                   const float* __restrict__ alpha, float* __restrict__ var, int N, int F) {
    __shared__ float sdata[256];
    const int f = blockIdx.x;
    const float am = alpha[f] * mean[f];
    float s = 0.0f;
    for (int r = threadIdx.x; r < N; r += 256) {
        float sh = x[(size_t)r * F + f] - am;
        s += sh * sh;
    }
    sdata[threadIdx.x] = s;
    __syncthreads();
    for (int off = 128; off > 0; off >>= 1) {
        if (threadIdx.x < off) sdata[threadIdx.x] += sdata[threadIdx.x + off];
        __syncthreads();
    }
    if (threadIdx.x == 0) var[f] = sdata[0] / (float)N;
}

__global__ void norm_apply_kernel(const float* __restrict__ x, const float* __restrict__ mean,
                                  const float* __restrict__ var, const float* __restrict__ gamma,
                                  const float* __restrict__ beta, const float* __restrict__ alpha,
                                  const float* __restrict__ prev, float* __restrict__ hnew,
                                  bf16_t* __restrict__ hbf, bf16_t* __restrict__ jk,
                                  int N, int F, int jkStride, int jkOff) {
    long idx = (long)blockIdx.x * blockDim.x + threadIdx.x;
    long total = (long)N * F;
    if (idx >= total) return;
    int n = (int)(idx / F), f = (int)(idx % F);
    float sh = x[idx] - alpha[f] * mean[f];
    float y  = gamma[f] * sh * rsqrtf(var[f] + 1e-5f) + beta[f];
    if (prev) y += prev[idx];
    hnew[idx] = y;
    hbf [idx] = (bf16_t)y;
    jk[(size_t)n * jkStride + jkOff + f] = (bf16_t)y;
}

// ---- z = normalize(tanh(mlp) + inz); build din = [inz | z] (bf16, padded) --
// zf has row stride 320 (266 valid cols), inzf stride 288, din stride 576.
__global__ __launch_bounds__(128)
void z_din_kernel(const float* __restrict__ zf, const float* __restrict__ inzf,
                  bf16_t* __restrict__ din, int N) {
    const int warp = threadIdx.x >> 5, lane = threadIdx.x & 31;
    const long r = (long)blockIdx.x * 4 + warp;
    if (r >= N) return;
    const float* zr = zf   + (size_t)r * 320;
    const float* ir = inzf + (size_t)r * 288;
    float ss = 0.0f;
    for (int c = lane; c < 266; c += 32) {
        float zt = zr[c] + ir[c];
        ss += zt * zt;
    }
    for (int off = 16; off > 0; off >>= 1) ss += __shfl_xor(ss, off, 32);
    const float inv = 1.0f / (sqrtf(ss) + 1e-10f);
    bf16_t* dr = din + (size_t)r * 576;
    for (int c = lane; c < 288; c += 32) {
        float iv = ir[c];
        float zt = (c < 266) ? (zr[c] + iv) * inv : 0.0f;
        dr[c]       = (bf16_t)iv;
        dr[288 + c] = (bf16_t)zt;
    }
}

// ---- log_softmax over 20 classes (padded logits stride 64), warp per row ---
__global__ __launch_bounds__(128)
void logsoftmax_kernel(const float* __restrict__ logits, float* __restrict__ out, int N) {
    const int warp = threadIdx.x >> 5, lane = threadIdx.x & 31;
    const long r = (long)blockIdx.x * 4 + warp;
    if (r >= N) return;
    float x = (lane < 20) ? logits[(size_t)r * 64 + lane] : -3.4028235e38f;
    float m = x;
    for (int off = 16; off > 0; off >>= 1) m = fmaxf(m, __shfl_xor(m, off, 32));
    float e = (lane < 20) ? expf(x - m) : 0.0f;
    float s = e;
    for (int off = 16; off > 0; off >>= 1) s += __shfl_xor(s, off, 32);
    if (lane < 20) out[(size_t)r * 20 + lane] = x - m - logf(s);
}

// ---- weight conversion f32[K,F] -> bf16[Kpad,Fpad], zero padded ------------
__global__ void convw_kernel(const float* __restrict__ src, bf16_t* __restrict__ dst,
                             int K, int F, int Kpad, int Fpad) {
    long idx = (long)blockIdx.x * blockDim.x + threadIdx.x;
    long total = (long)Kpad * Fpad;
    if (idx >= total) return;
    int kp = (int)(idx / Fpad), f = (int)(idx % Fpad);
    float v = (kp < K && f < F) ? src[(size_t)kp * F + f] : 0.0f;
    dst[idx] = (bf16_t)v;
}

// D1 [532,256] -> [576,256]: rows 0..265 -> 0..265, rows 266..531 -> 288..553
__global__ void convw_split_kernel(const float* __restrict__ src, bf16_t* __restrict__ dst,
                                   int K1, int K2, int F, int half, int Kpad, int Fpad) {
    long idx = (long)blockIdx.x * blockDim.x + threadIdx.x;
    long total = (long)Kpad * Fpad;
    if (idx >= total) return;
    int kp = (int)(idx / Fpad), f = (int)(idx % Fpad);
    float v = 0.0f;
    if (f < F) {
        if (kp < K1)                              v = src[(size_t)kp * F + f];
        else if (kp >= half && kp < half + K2)    v = src[(size_t)(K1 + kp - half) * F + f];
    }
    dst[idx] = (bf16_t)v;
}

// ---------------------------------------------------------------------------
// Host launcher
// ---------------------------------------------------------------------------
static inline long cdivl(long a, long b) { return (a + b - 1) / b; }

extern "C" void kernel_launch(void* const* d_in, const int* in_sizes, int n_in,
                              void* d_out, int out_size, void* d_ws, size_t ws_size,
                              hipStream_t stream) {
    const int N = in_sizes[0] / 10;
    const int E = in_sizes[2] / 2;

    // ---- inputs (pytree flatten order of setup_inputs dict) ----
    const float* x_res = (const float*)d_in[0];
    const float* pos   = (const float*)d_in[1];
    const int*   esrc  = (const int*)d_in[2];
    const int*   edst  = esrc + E;
    const float* Wl[3] = {(const float*)d_in[3], (const float*)d_in[6], (const float*)d_in[9]};
    const float* Wr[3] = {(const float*)d_in[4], (const float*)d_in[7], (const float*)d_in[10]};
    const float* sb[3] = {(const float*)d_in[5], (const float*)d_in[8], (const float*)d_in[11]};
    const float* gma[3] = {(const float*)d_in[12], (const float*)d_in[15], (const float*)d_in[18]};
    const float* bta[3] = {(const float*)d_in[13], (const float*)d_in[16], (const float*)d_in[19]};
    const float* alp[3] = {(const float*)d_in[14], (const float*)d_in[17], (const float*)d_in[20]};
    const float* W1 = (const float*)d_in[21]; const float* bl1 = (const float*)d_in[22];
    const float* W2 = (const float*)d_in[23]; const float* bl2 = (const float*)d_in[24];
    const float* W3 = (const float*)d_in[25]; const float* bl3 = (const float*)d_in[26];
    const float* D1 = (const float*)d_in[27]; const float* c1  = (const float*)d_in[28];
    const float* D2 = (const float*)d_in[29]; const float* c2  = (const float*)d_in[30];
    const float* D3 = (const float*)d_in[31]; const float* c3  = (const float*)d_in[32];

    // ---- workspace carve-out ----
    char* ws = (char*)d_ws;
    size_t off = 0;
    auto alloc = [&](size_t bytes) -> void* {
        void* p = ws + off;
        off += (bytes + 255) & ~(size_t)255;
        return p;
    };
    float*  inzf   = (float*) alloc((size_t)N * 288 * 4);
    bf16_t* hbf0   = (bf16_t*)alloc((size_t)N * 288 * 2);
    float*  deg    = (float*) alloc((size_t)N * 4);
    float*  aggrf  = (float*) alloc((size_t)N * 288 * 4);
    bf16_t* aggrbf = (bf16_t*)alloc((size_t)N * 288 * 2);
    float*  outf   = (float*) alloc((size_t)N * 128 * 4);
    float*  hA     = (float*) alloc((size_t)N * 128 * 4);
    float*  hB     = (float*) alloc((size_t)N * 128 * 4);
    bf16_t* hbfL   = (bf16_t*)alloc((size_t)N * 128 * 2);
    float*  meanv  = (float*) alloc(128 * 4);
    float*  varv   = (float*) alloc(128 * 4);
    bf16_t* jk     = (bf16_t*)alloc((size_t)N * 384 * 2);
    bf16_t* m1     = (bf16_t*)alloc((size_t)N * 256 * 2);
    bf16_t* m2     = (bf16_t*)alloc((size_t)N * 256 * 2);
    float*  zf     = (float*) alloc((size_t)N * 320 * 4);
    bf16_t* din    = (bf16_t*)alloc((size_t)N * 576 * 2);
    bf16_t* dd1    = (bf16_t*)alloc((size_t)N * 256 * 2);
    bf16_t* dd2    = (bf16_t*)alloc((size_t)N * 256 * 2);
    float*  logits = (float*) alloc((size_t)N * 64 * 4);
    bf16_t* Wl0b = (bf16_t*)alloc(288 * 128 * 2);
    bf16_t* Wr0b = (bf16_t*)alloc(288 * 128 * 2);
    bf16_t* Wl1b = (bf16_t*)alloc(128 * 128 * 2);
    bf16_t* Wr1b = (bf16_t*)alloc(128 * 128 * 2);
    bf16_t* Wl2b = (bf16_t*)alloc(128 * 128 * 2);
    bf16_t* Wr2b = (bf16_t*)alloc(128 * 128 * 2);
    bf16_t* W1b  = (bf16_t*)alloc(384 * 256 * 2);
    bf16_t* W2b  = (bf16_t*)alloc(256 * 256 * 2);
    bf16_t* W3b  = (bf16_t*)alloc(256 * 320 * 2);
    bf16_t* D1b  = (bf16_t*)alloc(576 * 256 * 2);
    bf16_t* D2b  = (bf16_t*)alloc(256 * 256 * 2);
    bf16_t* D3b  = (bf16_t*)alloc(256 * 64 * 2);

    const int BS = 256;
    auto g1 = [&](long total) { return dim3((unsigned)cdivl(total, BS)); };

    // ---- weight conversions (tiny) ----
    hipLaunchKernelGGL(convw_kernel, g1(288L*128), dim3(BS), 0, stream, Wl[0], Wl0b, 266, 128, 288, 128);
    hipLaunchKernelGGL(convw_kernel, g1(288L*128), dim3(BS), 0, stream, Wr[0], Wr0b, 266, 128, 288, 128);
    hipLaunchKernelGGL(convw_kernel, g1(128L*128), dim3(BS), 0, stream, Wl[1], Wl1b, 128, 128, 128, 128);
    hipLaunchKernelGGL(convw_kernel, g1(128L*128), dim3(BS), 0, stream, Wr[1], Wr1b, 128, 128, 128, 128);
    hipLaunchKernelGGL(convw_kernel, g1(128L*128), dim3(BS), 0, stream, Wl[2], Wl2b, 128, 128, 128, 128);
    hipLaunchKernelGGL(convw_kernel, g1(128L*128), dim3(BS), 0, stream, Wr[2], Wr2b, 128, 128, 128, 128);
    hipLaunchKernelGGL(convw_kernel, g1(384L*256), dim3(BS), 0, stream, W1, W1b, 384, 256, 384, 256);
    hipLaunchKernelGGL(convw_kernel, g1(256L*256), dim3(BS), 0, stream, W2, W2b, 256, 256, 256, 256);
    hipLaunchKernelGGL(convw_kernel, g1(256L*320), dim3(BS), 0, stream, W3, W3b, 256, 266, 256, 320);
    hipLaunchKernelGGL(convw_split_kernel, g1(576L*256), dim3(BS), 0, stream, D1, D1b, 266, 266, 256, 288, 576, 256);
    hipLaunchKernelGGL(convw_kernel, g1(256L*256), dim3(BS), 0, stream, D2, D2b, 256, 256, 256, 256);
    hipLaunchKernelGGL(convw_kernel, g1(256L*64),  dim3(BS), 0, stream, D3, D3b, 256, 20, 256, 64);

    // ---- h0 = [x_res | positions], padded; degrees ----
    hipLaunchKernelGGL(concat_kernel, g1((long)N*288), dim3(BS), 0, stream, x_res, pos, inzf, hbf0, N);
    hipLaunchKernelGGL(zero_f32_kernel, g1(N), dim3(BS), 0, stream, deg, (long)N);
    hipLaunchKernelGGL(degree_kernel, g1(E), dim3(BS), 0, stream, edst, deg, E);

    // ---- SAGE layers ----
    struct { const float* h; int Kin, Kpad; const bf16_t *Awl, *Awr, *hb; const float* prev;
             float* hout; int jkOff; int li; } L[3] = {
        { inzf, 266, 288, Wl0b, Wr0b, hbf0, nullptr, hA, 0,   0 },
        { hA,   128, 128, Wl1b, Wr1b, hbfL, hA,      hB, 128, 1 },
        { hB,   128, 128, Wl2b, Wr2b, hbfL, hB,      hA, 256, 2 },
    };
    for (int i = 0; i < 3; ++i) {
        const int Kin = L[i].Kin, Kpad = L[i].Kpad, li = L[i].li;
        hipLaunchKernelGGL(zero_f32_kernel, g1((long)N*Kpad), dim3(BS), 0, stream, aggrf, (long)N*Kpad);
        hipLaunchKernelGGL(scatter_add_kernel, g1((long)E*Kin), dim3(BS), 0, stream,
                           L[i].h, esrc, edst, aggrf, (long)E*Kin, Kin, Kpad, Kpad);
        hipLaunchKernelGGL(aggr_finalize_kernel, g1((long)N*Kpad), dim3(BS), 0, stream,
                           aggrf, deg, aggrbf, N, Kin, Kpad);
        // out = gelu(aggr@Wl + h@Wr + b)   (F=128 -> grid.y = 2)
        hipLaunchKernelGGL(HIP_KERNEL_NAME(wmma_gemm_tpl<true>),
                           dim3((unsigned)cdivl(N,128), 2), dim3(256), 0, stream,
                           aggrbf, L[i].hb, L[i].Awl, L[i].Awr, sb[li], 128,
                           outf, (bf16_t*)nullptr, N, Kpad, 128, 1);
        // GraphNorm + residual, emit f32 / bf16 / JK slice
        hipLaunchKernelGGL(colsum_kernel, dim3(128), dim3(256), 0, stream, outf, meanv, N, 128);
        hipLaunchKernelGGL(colvar_kernel, dim3(128), dim3(256), 0, stream, outf, meanv, alp[li], varv, N, 128);
        hipLaunchKernelGGL(norm_apply_kernel, g1((long)N*128), dim3(BS), 0, stream,
                           outf, meanv, varv, gma[li], bta[li], alp[li],
                           L[i].prev, L[i].hout, hbfL, jk, N, 128, 384, L[i].jkOff);
    }

    // ---- lin MLP: jk(384) -> 256 -> 256 -> 320(tanh) ----
    hipLaunchKernelGGL(HIP_KERNEL_NAME(wmma_gemm_tpl<false>),
                       dim3((unsigned)cdivl(N,128), 4), dim3(256), 0, stream,
                       jk, (const bf16_t*)nullptr, W1b, (const bf16_t*)nullptr, bl1, 256,
                       (float*)nullptr, m1, N, 384, 256, 1);
    hipLaunchKernelGGL(HIP_KERNEL_NAME(wmma_gemm_tpl<false>),
                       dim3((unsigned)cdivl(N,128), 4), dim3(256), 0, stream,
                       m1, (const bf16_t*)nullptr, W2b, (const bf16_t*)nullptr, bl2, 256,
                       (float*)nullptr, m2, N, 256, 256, 1);
    hipLaunchKernelGGL(HIP_KERNEL_NAME(wmma_gemm_tpl<false>),
                       dim3((unsigned)cdivl(N,128), 5), dim3(256), 0, stream,
                       m2, (const bf16_t*)nullptr, W3b, (const bf16_t*)nullptr, bl3, 266,
                       zf, (bf16_t*)nullptr, N, 256, 320, 2);
    // z = normalize(tanh + inz); din = [inz | z]
    hipLaunchKernelGGL(z_din_kernel, dim3((unsigned)cdivl(N,4)), dim3(128), 0, stream, zf, inzf, din, N);

    // ---- decoder: din(576) -> 256 -> 256 -> 64(logits) ----
    hipLaunchKernelGGL(HIP_KERNEL_NAME(wmma_gemm_tpl<false>),
                       dim3((unsigned)cdivl(N,128), 4), dim3(256), 0, stream,
                       din, (const bf16_t*)nullptr, D1b, (const bf16_t*)nullptr, c1, 256,
                       (float*)nullptr, dd1, N, 576, 256, 1);
    hipLaunchKernelGGL(HIP_KERNEL_NAME(wmma_gemm_tpl<false>),
                       dim3((unsigned)cdivl(N,128), 4), dim3(256), 0, stream,
                       dd1, (const bf16_t*)nullptr, D2b, (const bf16_t*)nullptr, c2, 256,
                       (float*)nullptr, dd2, N, 256, 256, 1);
    hipLaunchKernelGGL(HIP_KERNEL_NAME(wmma_gemm_tpl<false>),
                       dim3((unsigned)cdivl(N,128), 1), dim3(256), 0, stream,
                       dd2, (const bf16_t*)nullptr, D3b, (const bf16_t*)nullptr, c3, 20,
                       logits, (bf16_t*)nullptr, N, 256, 64, 0);

    // ---- log_softmax -> d_out [N,20] f32 ----
    hipLaunchKernelGGL(logsoftmax_kernel, dim3((unsigned)cdivl(N,4)), dim3(128), 0, stream,
                       logits, (float*)d_out, N);
}
